// CrossAttention_3908420240258
// MI455X (gfx1250) — 16-bit and 8-bit matrix operand order repaired and checked by reading, compiled, not run on hardware
//
#include <hip/hip_runtime.h>
#include <hip/hip_bf16.h>

typedef __attribute__((ext_vector_type(16))) _Float16 v16h;
typedef __attribute__((ext_vector_type(8)))  _Float16 v8h;
typedef __attribute__((ext_vector_type(8)))  float    v8f;
typedef __attribute__((ext_vector_type(8)))  int      v8i;

#define EPSQ 1e-5f
#define DIMK 1024
#define ROWSA 4096
#define HEADS 16
#define HD 64
#define SEQ 2048

__device__ __forceinline__ unsigned lds_addr_of(const void* p) {
  return (unsigned)(unsigned long long)p;
}
__device__ __forceinline__ void async_ld_b128(unsigned lds_byte_addr, const void* gaddr) {
  asm volatile("global_load_async_to_lds_b128 %0, %1, off"
               :: "v"(lds_byte_addr), "v"(gaddr) : "memory");
}
__device__ __forceinline__ void wait_async_all() {
  asm volatile("s_wait_asynccnt 0x0" ::: "memory");
}

__global__ __launch_bounds__(256) void k_wsum_partial(const float* __restrict__ w,
                                                      float* __restrict__ part) {
  __shared__ float red[256];
  const int tid = threadIdx.x;
  const size_t base = (size_t)blockIdx.x * 4096 + tid;
  float s = 0.f;
#pragma unroll
  for (int i = 0; i < 16; ++i) s += fabsf(w[base + (size_t)i * 256]);
  red[tid] = s; __syncthreads();
  for (int off = 128; off > 0; off >>= 1) {
    if (tid < off) red[tid] += red[tid + off];
    __syncthreads();
  }
  if (tid == 0) part[blockIdx.x] = red[0];
}

__global__ __launch_bounds__(256) void k_wsum_final(const float* __restrict__ part,
                                                    float* __restrict__ wmean) {
  __shared__ float red[256];
  const int tid = threadIdx.x;
  red[tid] = part[blockIdx.x * 256 + tid]; __syncthreads();
  for (int off = 128; off > 0; off >>= 1) {
    if (tid < off) red[tid] += red[tid + off];
    __syncthreads();
  }
  if (tid == 0) wmean[blockIdx.x] = fmaxf(red[0] * (1.0f / (1024.f * 1024.f)), EPSQ);
}

__global__ __launch_bounds__(256) void k_wquant(const float* __restrict__ w,
                                                const float* __restrict__ wmean, int sidx,
                                                signed char* __restrict__ wq) {
  const float scale = 1.0f / wmean[sidx];
  const size_t base = ((size_t)blockIdx.x * 256 + threadIdx.x) * 4;
  float4 v = *(const float4*)(w + base);
  char4 o;
  o.x = (signed char)fminf(fmaxf(rintf(v.x * scale), -1.f), 1.f);
  o.y = (signed char)fminf(fmaxf(rintf(v.y * scale), -1.f), 1.f);
  o.z = (signed char)fminf(fmaxf(rintf(v.z * scale), -1.f), 1.f);
  o.w = (signed char)fminf(fmaxf(rintf(v.w * scale), -1.f), 1.f);
  *(char4*)(wq + base) = o;
}

__global__ __launch_bounds__(256) void k_aquant(const float* __restrict__ x,
                                                signed char* __restrict__ xq,
                                                float* __restrict__ dq) {
  __shared__ float red[256];
  const int tid = threadIdx.x;
  const size_t row = blockIdx.x;
  const float* xr = x + row * 1024;
  float vals[4]; float m = 0.f;
#pragma unroll
  for (int i = 0; i < 4; ++i) { vals[i] = xr[tid + i * 256]; m = fmaxf(m, fabsf(vals[i])); }
  red[tid] = m; __syncthreads();
  for (int off = 128; off > 0; off >>= 1) {
    if (tid < off) red[tid] = fmaxf(red[tid], red[tid + off]);
    __syncthreads();
  }
  const float mx = fmaxf(red[0], EPSQ);
  const float scale = 127.f / mx;
#pragma unroll
  for (int i = 0; i < 4; ++i) {
    float qv = fminf(fmaxf(rintf(vals[i] * scale), -128.f), 127.f);
    xq[row * 1024 + tid + i * 256] = (signed char)qv;
  }
  if (tid == 0) dq[row] = mx * (1.f / 127.f);
}

__global__ __launch_bounds__(256) void k_gemm_i8(const signed char* __restrict__ A,
                                                 const signed char* __restrict__ W,
                                                 const float* __restrict__ dqa,
                                                 const float* __restrict__ wmean, int sidx,
                                                 void* __restrict__ out, int mode, float escale) {
  __shared__ __align__(16) signed char wbuf[2][64][80];

  const int tid  = threadIdx.x;
  const int lane = tid & 31;
  const int wave = tid >> 5;
  const int l16  = lane & 15;
  const int half = lane >> 4;
  const int row0 = blockIdx.x * 128 + wave * 16;
  const int col0 = blockIdx.y * 64;

  const int wrow = tid >> 2;
  const int win  = (tid & 3) * 16;
  const signed char* wsrc_row = W + (size_t)(col0 + wrow) * 1024 + win;

  v8i acc[4];
#pragma unroll
  for (int c = 0; c < 4; ++c)
#pragma unroll
    for (int e = 0; e < 8; ++e) acc[c][e] = 0;

  const signed char* arow = A + (size_t)(row0 + l16) * 1024;

  async_ld_b128(lds_addr_of(&wbuf[0][wrow][win]), wsrc_row);

#pragma unroll 1
  for (int k0 = 0; k0 < 1024; k0 += 64) {
    const int cur = (k0 >> 6) & 1;

    v8i a;
#pragma unroll
    for (int g = 0; g < 4; ++g) {
      int2 d = *(const int2*)(arow + k0 + (g >> 1) * 32 + half * 16 + (g & 1) * 8);
      a[2 * g] = d.x; a[2 * g + 1] = d.y;
    }

    wait_async_all();
    __syncthreads();
    if (k0 + 64 < 1024)
      async_ld_b128(lds_addr_of(&wbuf[cur ^ 1][wrow][win]), wsrc_row + k0 + 64);

#pragma unroll
    for (int c = 0; c < 4; ++c) {
      const signed char* brow = &wbuf[cur][c * 16 + l16][0];
      int4 b0 = *(const int4*)(brow + half * 16);
      int4 b1 = *(const int4*)(brow + 32 + half * 16);
      v8i b;
      b[0] = b0.x; b[1] = b0.y; b[2] = b0.z; b[3] = b0.w;
      b[4] = b1.x; b[5] = b1.y; b[6] = b1.z; b[7] = b1.w;
      acc[c] = __builtin_amdgcn_wmma_i32_16x16x64_iu8(true, a, true, b, acc[c], false, false);
    }
  }

  const float wsc = wmean[sidx] * escale;
#pragma unroll
  for (int c = 0; c < 4; ++c) {
#pragma unroll
    for (int g = 0; g < 8; ++g) {
      const int row = row0 + g + 8 * half;
      const int col = col0 + c * 16 + l16;
      const float val = (float)acc[c][g] * dqa[row] * wsc;
      if (mode == 3) {
        ((float*)out)[(size_t)row * 1024 + col] = val;
      } else {
        const int b_i = row >> 11, nn = row & 2047;
        const int h = col >> 6,  dd = col & 63;
        const _Float16 hv = (_Float16)val;
        if (mode == 2)
          ((_Float16*)out)[(((size_t)b_i * HEADS + h) * HD + dd) * SEQ + nn] = hv;
        else
          ((_Float16*)out)[(((size_t)b_i * HEADS + h) * SEQ + nn) * HD + dd] = hv;
      }
    }
  }
}

__global__ __launch_bounds__(256) void k_attn(const _Float16* __restrict__ q,
                                              const _Float16* __restrict__ k,
                                              const _Float16* __restrict__ vt,
                                              float* __restrict__ ao) {
  __shared__ __align__(16) _Float16 kbuf[2][32][72];
  __shared__ __align__(16) _Float16 vbuf[2][64][40];
  __shared__ _Float16 pbuf[8][16][34];

  const int tid  = threadIdx.x;
  const int lane = tid & 31;
  const int wave = tid >> 5;
  const int l16  = lane & 15;
  const int half = lane >> 4;
  const int bh   = blockIdx.y;
  const int b_i  = bh >> 4, h = bh & 15;
  const int n0   = blockIdx.x * 128 + wave * 16;

  const _Float16* qh = q  + (size_t)bh * SEQ * HD;
  const _Float16* kh = k  + (size_t)bh * SEQ * HD;
  const _Float16* vh = vt + (size_t)bh * HD * SEQ;

  const int krow_t = tid >> 3, kin_t = (tid & 7) * 16;
  const int vrow_t = tid >> 2, vin_t = (tid & 3) * 16;

  auto issue_tiles = [&](int buf, int mt) {
    const char* gk = (const char*)kh + (size_t)(mt + krow_t) * 128 + kin_t;
    async_ld_b128(lds_addr_of((const char*)&kbuf[buf][krow_t][0] + kin_t), gk);
    const char* gv = (const char*)vh + (size_t)vrow_t * (SEQ * 2) + mt * 2 + vin_t;
    async_ld_b128(lds_addr_of((const char*)&vbuf[buf][vrow_t][0] + vin_t), gv);
  };

  v16h aq[2];
  {
    const _Float16* qrow = qh + (size_t)(n0 + l16) * HD;
#pragma unroll
    for (int kk = 0; kk < 2; ++kk) {
      v8h lo = *(const v8h*)(qrow + kk * 32 + half * 8);
      v8h hi = *(const v8h*)(qrow + kk * 32 + 16 + half * 8);
#pragma unroll
      for (int e = 0; e < 8; ++e) { aq[kk][e] = lo[e]; aq[kk][8 + e] = hi[e]; }
    }
  }

  v8f acc[4];
#pragma unroll
  for (int c = 0; c < 4; ++c)
#pragma unroll
    for (int e = 0; e < 8; ++e) acc[c][e] = 0.f;
  float run_max[8], run_sum[8];
#pragma unroll
  for (int g = 0; g < 8; ++g) { run_max[g] = -3.0e38f; run_sum[g] = 0.f; }

  issue_tiles(0, 0);

#pragma unroll 1
  for (int mt = 0; mt < SEQ; mt += 32) {
    const int cur = (mt >> 5) & 1;
    wait_async_all();
    __syncthreads();
    if (mt + 32 < SEQ) issue_tiles(cur ^ 1, mt + 32);

    v8f s[2];
#pragma unroll
    for (int t = 0; t < 2; ++t) {
#pragma unroll
      for (int e = 0; e < 8; ++e) s[t][e] = 0.f;
      const _Float16* krow = &kbuf[cur][t * 16 + l16][0];
#pragma unroll
      for (int kk = 0; kk < 2; ++kk) {
        v8h lo = *(const v8h*)(krow + kk * 32 + half * 8);
        v8h hi = *(const v8h*)(krow + kk * 32 + 16 + half * 8);
        v16h bk;
#pragma unroll
        for (int e = 0; e < 8; ++e) { bk[e] = lo[e]; bk[8 + e] = hi[e]; }
        s[t] = __builtin_amdgcn_wmma_f32_16x16x32_f16(false, aq[kk], false, bk,
                                                      (short)0, s[t], false, false);
      }
    }
    float tmax[8];
#pragma unroll
    for (int g = 0; g < 8; ++g) tmax[g] = fmaxf(s[0][g], s[1][g]);
#pragma unroll
    for (int off = 1; off < 16; off <<= 1)
#pragma unroll
      for (int g = 0; g < 8; ++g) tmax[g] = fmaxf(tmax[g], __shfl_xor(tmax[g], off, 32));

    float p0[8], p1[8], corr[8], rsum[8];
#pragma unroll
    for (int g = 0; g < 8; ++g) {
      const float nm = fmaxf(run_max[g], tmax[g]);
      corr[g] = __expf(run_max[g] - nm);
      run_max[g] = nm;
      p0[g] = __expf(s[0][g] - nm);
      p1[g] = __expf(s[1][g] - nm);
      rsum[g] = p0[g] + p1[g];
    }
#pragma unroll
    for (int off = 1; off < 16; off <<= 1)
#pragma unroll
      for (int g = 0; g < 8; ++g) rsum[g] += __shfl_xor(rsum[g], off, 32);
#pragma unroll
    for (int g = 0; g < 8; ++g) run_sum[g] = run_sum[g] * corr[g] + rsum[g];
#pragma unroll
    for (int c = 0; c < 4; ++c)
#pragma unroll
      for (int g = 0; g < 8; ++g) acc[c][g] *= corr[g];

    __syncthreads();
#pragma unroll
    for (int g = 0; g < 8; ++g) {
      pbuf[wave][g + 8 * half][l16]      = (_Float16)p0[g];
      pbuf[wave][g + 8 * half][l16 + 16] = (_Float16)p1[g];
    }
    __syncthreads();
    v16h ap;
#pragma unroll
    for (int e = 0; e < 8; ++e) {
      ap[e]     = pbuf[wave][l16][8 * half + e];
      ap[8 + e] = pbuf[wave][l16][16 + 8 * half + e];
    }
#pragma unroll
    for (int c = 0; c < 4; ++c) {
      const _Float16* vrow = &vbuf[cur][c * 16 + l16][0];
      v8h lo = *(const v8h*)(vrow + half * 8);
      v8h hi = *(const v8h*)(vrow + 16 + half * 8);
      v16h bv;
#pragma unroll
      for (int e = 0; e < 8; ++e) { bv[e] = lo[e]; bv[8 + e] = hi[e]; }
      acc[c] = __builtin_amdgcn_wmma_f32_16x16x32_f16(false, ap, false, bv,
                                                      (short)0, acc[c], false, false);
    }
  }

#pragma unroll
  for (int c = 0; c < 4; ++c)
#pragma unroll
    for (int g = 0; g < 8; ++g) {
      const int nn = n0 + g + 8 * half;
      const float val = acc[c][g] / run_sum[g];
      ao[((size_t)b_i * SEQ + nn) * 1024 + h * 64 + c * 16 + l16] = val;
    }
}

extern "C" void kernel_launch(void* const* d_in, const int* in_sizes, int n_in,
                              void* d_out, int out_size, void* d_ws, size_t ws_size,
                              hipStream_t stream) {
  (void)in_sizes; (void)n_in; (void)out_size; (void)ws_size;
  const float* x   = (const float*)d_in[0];
  const float* ctx = (const float*)d_in[1];
  const float* wsrc[4] = {(const float*)d_in[2], (const float*)d_in[3],
                          (const float*)d_in[4], (const float*)d_in[5]};
  char* ws = (char*)d_ws;
  const size_t MB = 1024ull * 1024ull;

  signed char* wqi[4];
  for (int i = 0; i < 4; ++i) wqi[i] = (signed char*)(ws + i * MB);
  signed char* xq  = (signed char*)(ws + 4 * MB);
  signed char* cq  = (signed char*)(ws + 8 * MB);
  signed char* aoq = (signed char*)(ws + 12 * MB);
  _Float16* qf  = (_Float16*)(ws + 16 * MB);
  _Float16* kf  = (_Float16*)(ws + 24 * MB);
  _Float16* vtf = (_Float16*)(ws + 32 * MB);
  float* ao     = (float*)(ws + 40 * MB);
  float* dqx    = (float*)(ws + 56 * MB);
  float* dqc    = dqx + ROWSA;
  float* dqao   = dqc + ROWSA;
  float* wmean  = dqao + ROWSA;
  float* wpart  = wmean + 16;

  for (int i = 0; i < 4; ++i)
    k_wsum_partial<<<256, 256, 0, stream>>>(wsrc[i], wpart + i * 256);
  k_wsum_final<<<4, 256, 0, stream>>>(wpart, wmean);

  for (int i = 0; i < 4; ++i)
    k_wquant<<<1024, 256, 0, stream>>>(wsrc[i], wmean, i, wqi[i]);

  k_aquant<<<ROWSA, 256, 0, stream>>>(x,   xq, dqx);
  k_aquant<<<ROWSA, 256, 0, stream>>>(ctx, cq, dqc);

  dim3 gg(ROWSA / 128, DIMK / 64);
  k_gemm_i8<<<gg, 256, 0, stream>>>(xq, wqi[0], dqx, wmean, 0, (void*)qf, 0, 0.125f);
  k_gemm_i8<<<gg, 256, 0, stream>>>(cq, wqi[1], dqc, wmean, 1, (void*)kf, 1, 1.0f);
  k_gemm_i8<<<gg, 256, 0, stream>>>(cq, wqi[2], dqc, wmean, 2, (void*)vtf, 2, 1.0f);

  k_attn<<<dim3(SEQ / 128, 32), 256, 0, stream>>>(qf, kf, vtf, ao);

  k_aquant<<<ROWSA, 256, 0, stream>>>(ao, aoq, dqao);
  k_gemm_i8<<<gg, 256, 0, stream>>>(aoq, wqi[3], dqao, wmean, 3, d_out, 3, 1.0f);
}
